// PAO_model_50835232916290
// MI455X (gfx1250) — matrix operand order repaired and checked by reading, not compiled, not run on hardware
//
#include <hip/hip_runtime.h>
#include <math.h>

typedef __attribute__((ext_vector_type(16))) _Float16 v16h;
typedef __attribute__((ext_vector_type(8)))  float    v8f;

#define BATCH 32768

__global__ __launch_bounds__(256) void k_edge(
    const float* __restrict__ pos, const float* __restrict__ xc,
    const float* __restrict__ zin, const float* __restrict__ fw1,
    const float* __restrict__ fw2, float* __restrict__ aBuf)
{
  // per-wave LDS regions (8 waves / block)
  __shared__ _Float16 ldsH[8][16][66];   // hact transpose buffer [row m][col h]
  __shared__ float    ldsT[8][32][21];   // tpw per edge [n][20]

  const int lane = threadIdx.x & 31;
  const int wv   = threadIdx.x >> 5;
  const int b    = blockIdx.x * 8 + wv;
  const int hi   = lane >> 4;        // 0: lanes 0-15, 1: lanes 16-31
  const int col  = lane & 15;

  const float INV_SQRT10 = 0.31622776601683794f;
  const float SQRT10     = 3.1622776601683795f;
  const float SILU_CST   = 1.6790503f;   // normalize2mom(silu) under N(0,1)

  // ---- layer-1 B operand: W1[k, 16t+col]/sqrt(10). B layout: K = q + 16*hi
  v16h B1[4];
  #pragma unroll
  for (int t = 0; t < 4; ++t) {
    #pragma unroll
    for (int q = 0; q < 16; ++q) {
      float w = (hi == 0 && q < 10) ? fw1[q * 64 + t * 16 + col] * INV_SQRT10 : 0.f;
      B1[t][q] = (_Float16)w;
    }
  }
  // ---- layer-2 B operand: W2[k, 16nt+col]/8. K = 32*kc + q + 16*hi
  v16h B2[2][2];
  #pragma unroll
  for (int kc = 0; kc < 2; ++kc) {
    #pragma unroll
    for (int nt = 0; nt < 2; ++nt) {
      #pragma unroll
      for (int q = 0; q < 16; ++q) {
        int k  = 32 * kc + q + (hi ? 16 : 0);
        int tc = nt * 16 + col;
        float w = (tc < 20) ? fw2[k * 20 + tc] * 0.125f : 0.f;
        B2[kc][nt][q] = (_Float16)w;
      }
    }
  }

  float vs0[3], vs1[3], zs0[2], zs1[2];   // saved geometry per tile pass

  #pragma unroll
  for (int T = 0; T < 2; ++T) {
    // lanes L and L+16 both own edge m=col of this tile (A-matrix row M=col)
    const int e  = b * 32 + T * 16 + col;
    float ex = pos[e * 3 + 0] - xc[b * 3 + 0];
    float ey = pos[e * 3 + 1] - xc[b * 3 + 1];
    float ez = pos[e * 3 + 2] - xc[b * 3 + 2];
    float r   = sqrtf(ex * ex + ey * ey + ez * ez);
    float inv = __builtin_amdgcn_rcpf(fmaxf(r, 1e-9f));   // v_rcp_f32
    float vx = ex * inv, vy = ey * inv, vz = ez * inv;
    float z0 = zin[e * 2 + 0], z1 = zin[e * 2 + 1];
    if (T == 0) { vs0[0]=vx; vs0[1]=vy; vs0[2]=vz; zs0[0]=z0; zs0[1]=z1; }
    else        { vs1[0]=vx; vs1[1]=vy; vs1[2]=vz; zs1[0]=z0; zs1[1]=z1; }

    // radial basis: emb[k] = cos(pi/2 * diff)*sqrt(10), diff = 2.2 r - (k+1)
    float emb[10];
    #pragma unroll
    for (int k = 0; k < 10; ++k) {
      float diff = r * 2.2f - (float)(k + 1);
      float c = __cosf(1.5707963267948966f * diff) * SQRT10;
      emb[k] = (diff > -1.f && diff < 1.f) ? c : 0.f;
    }

    // A operand (16x32 f16): K = q + 16*hi, zero-padded past K=9
    v16h A1;
    #pragma unroll
    for (int q = 0; q < 16; ++q) {
      const int klo = q;
      const int khi = klo + 16;
      float lo = (klo < 10) ? emb[klo] : 0.f;
      float hv = (khi < 10) ? emb[khi] : 0.f;
      A1[q] = (_Float16)(hi ? hv : lo);
    }

    // ---- layer 1: 4 WMMAs over the 64 hidden units
    v8f o1[4];
    #pragma unroll
    for (int t = 0; t < 4; ++t) {
      v8f c = {};
      o1[t] = __builtin_amdgcn_wmma_f32_16x16x32_f16(
          false, A1, false, B1[t], (short)0, c, false, false);
    }

    // silu + scale, stash to LDS in [edge row][h] layout (D rows M = r + 8*hi)
    #pragma unroll
    for (int t = 0; t < 4; ++t) {
      #pragma unroll
      for (int rr = 0; rr < 8; ++rr) {
        float o = o1[t][rr];
        float sg = __builtin_amdgcn_rcpf(1.f + __expf(-o));  // fast sigmoid
        float hact = SILU_CST * o * sg;
        ldsH[wv][rr + hi * 8][t * 16 + col] = (_Float16)hact;
      }
    }
    __syncthreads();

    // re-read as A operand of layer 2: row m=col, K = 32kc + pattern
    v16h A2[2];
    #pragma unroll
    for (int kc = 0; kc < 2; ++kc) {
      #pragma unroll
      for (int q = 0; q < 16; ++q) {
        const int klo = 32 * kc + q;
        const int khi = klo + 16;
        _Float16 lo = ldsH[wv][col][klo];
        _Float16 hv = ldsH[wv][col][khi];
        A2[kc][q] = hi ? hv : lo;
      }
    }

    // ---- layer 2: 2 K-chunks x 2 N-tiles
    v8f o2[2];
    #pragma unroll
    for (int nt = 0; nt < 2; ++nt) {
      v8f c = {};
      c = __builtin_amdgcn_wmma_f32_16x16x32_f16(
          false, A2[0], false, B2[0][nt], (short)0, c, false, false);
      c = __builtin_amdgcn_wmma_f32_16x16x32_f16(
          false, A2[1], false, B2[1][nt], (short)0, c, false, false);
      o2[nt] = c;
    }
    __syncthreads();   // protect ldsH before next tile overwrites it

    // store tpw[edge][0..19]
    #pragma unroll
    for (int rr = 0; rr < 8; ++rr) {
      int row = T * 16 + rr + hi * 8;
      ldsT[wv][row][col] = o2[0][rr];
      if (col < 4) ldsT[wv][row][16 + col] = o2[1][rr];
    }
  }
  __syncthreads();

  // ---- per-edge contributions; lane L owns edge n = L (tile = hi, m = col)
  float vxl = hi ? vs1[0] : vs0[0];
  float vyl = hi ? vs1[1] : vs0[1];
  float vzl = hi ? vs1[2] : vs0[2];
  float z0l = hi ? zs1[0] : zs0[0];
  float z1l = hi ? zs1[1] : zs0[1];

  float t20[20];
  #pragma unroll
  for (int c = 0; c < 20; ++c) t20[c] = ldsT[wv][lane][c];

  float contrib[20];
  #pragma unroll
  for (int w = 0; w < 5; ++w) {
    contrib[w] = 0.7071067811865476f * (t20[w] * z0l + t20[5 + w] * z1l);
    float g = 1.224744871391589f * (t20[10 + w] * z0l + t20[15 + w] * z1l);
    contrib[5 + w * 3 + 0] = g * vxl;
    contrib[5 + w * 3 + 1] = g * vyl;
    contrib[5 + w * 3 + 2] = g * vzl;
  }
  // wave reduction over the 32 edges
  #pragma unroll
  for (int j = 0; j < 20; ++j) {
    float v = contrib[j];
    #pragma unroll
    for (int off = 16; off > 0; off >>= 1) v += __shfl_xor(v, off, 32);
    contrib[j] = v;
  }
  if (lane == 0) {
    #pragma unroll
    for (int j = 0; j < 20; ++j)
      aBuf[b * 20 + j] = contrib[j] * 0.17677669529663687f;   // 1/sqrt(32)
  }
}

// ---------------- per-batch contractions + 8x8 Jacobi eigensolver ----------
__global__ __launch_bounds__(64) void k_batch(
    const float* __restrict__ aBuf,
    const float* __restrict__ w00, const float* __restrict__ w11,
    const float* __restrict__ w01, const float* __restrict__ w10,
    const float* __restrict__ w2,  float* __restrict__ out)
{
  const int b = blockIdx.x * 64 + threadIdx.x;
  if (b >= BATCH) return;

  const float S2 = 0.7071067811865476f;   // 1/sqrt(2)
  const float S6 = 0.4082482904638631f;   // 1/sqrt(6)
  const float C2 = 0.16329931618554522f;  // 2/(5*sqrt(6))

  float a0[5], a1[5][3];
  #pragma unroll
  for (int u = 0; u < 5; ++u) a0[u] = aBuf[b * 20 + u];
  #pragma unroll
  for (int u = 0; u < 5; ++u)
    #pragma unroll
    for (int i = 0; i < 3; ++i) a1[u][i] = aBuf[b * 20 + 5 + u * 3 + i];

  float dot[5][5];
  #pragma unroll
  for (int u = 0; u < 5; ++u)
    #pragma unroll
    for (int v = 0; v < 5; ++v)
      dot[u][v] = a1[u][0]*a1[v][0] + a1[u][1]*a1[v][1] + a1[u][2]*a1[v][2];

  // p0[6]
  float p0[6];
  #pragma unroll
  for (int s = 0; s < 6; ++s) {
    float acc = 0.f;
    #pragma unroll
    for (int u = 0; u < 5; ++u)
      #pragma unroll
      for (int v = 0; v < 5; ++v)
        acc += w00[(s*5+u)*5+v] * a0[u]*a0[v]
             + w11[(s*5+u)*5+v] * dot[u][v] * 0.5773502691896258f;
    p0[s] = 0.1414213562373095f * acc;     // sqrt(1/50)
  }
  // p1[4][3], then rotate by D1 (cyclic permutation)
  float p1[4][3];
  #pragma unroll
  for (int s = 0; s < 4; ++s)
    #pragma unroll
    for (int i = 0; i < 3; ++i) {
      float acc = 0.f;
      #pragma unroll
      for (int u = 0; u < 5; ++u)
        #pragma unroll
        for (int v = 0; v < 5; ++v)
          acc += w01[(s*5+u)*5+v] * a0[u]*a1[v][i]
               + w10[(s*5+u)*5+v] * a1[u][i]*a0[v];
      p1[s][i] = 0.1414213562373095f * acc;  // sqrt(3/50)/sqrt(3)
    }
  float p1r[4][3];
  #pragma unroll
  for (int s = 0; s < 4; ++s) {
    p1r[s][0] = p1[s][1]; p1r[s][1] = p1[s][2]; p1r[s][2] = p1[s][0];
  }
  // p2[3][5] via M2 bilinear forms, then rotate by D2
  float p2[3][5];
  #pragma unroll
  for (int s = 0; s < 3; ++s) {
    float q0=0,q1=0,q2=0,q3=0,q4=0;
    #pragma unroll
    for (int u = 0; u < 5; ++u)
      #pragma unroll
      for (int v = 0; v < 5; ++v) {
        float w = w2[(s*5+u)*5+v];
        float x0=a1[u][0], x1=a1[u][1], x2=a1[u][2];
        float y0=a1[v][0], y1=a1[v][1], y2=a1[v][2];
        q0 += w * S2 * (x0*y2 + x2*y0);
        q1 += w * S2 * (x0*y1 + x1*y0);
        q2 += w * S6 * (-x0*y0 + 2.f*x1*y1 - x2*y2);
        q3 += w * S2 * (x1*y2 + x2*y1);
        q4 += w * S2 * (-x0*y0 + x2*y2);
      }
    p2[s][0]=0.2f*q0; p2[s][1]=0.2f*q1; p2[s][2]=0.2f*q2;
    p2[s][3]=0.2f*q3; p2[s][4]=0.2f*q4;
  }
  float p2r[3][5];
  #pragma unroll
  for (int s = 0; s < 3; ++s) {
    p2r[s][0] = p2[s][1];
    p2r[s][1] = p2[s][3];
    p2r[s][2] = -0.5f*p2[s][2] + 0.8660254037844386f*p2[s][4];
    p2r[s][3] = p2[s][0];
    p2r[s][4] = -0.8660254037844386f*p2[s][2] - 0.5f*p2[s][4];
  }

  // assemble symmetric aux 8x8
  float A[8][8];
  #pragma unroll
  for (int i = 0; i < 8; ++i)
    #pragma unroll
    for (int j = 0; j < 8; ++j) A[i][j] = 0.f;

  A[0][0] += p0[0];
  A[0][1] += p0[1]; A[1][0] += p0[1];
  #pragma unroll
  for (int j = 0; j < 3; ++j) {
    A[0][2+j] += p1r[0][j]/3.f; A[2+j][0] += p1r[0][j]/3.f;
    A[0][5+j] += p1r[1][j]/3.f; A[5+j][0] += p1r[1][j]/3.f;
    A[1][2+j] += p1r[2][j]/3.f; A[2+j][1] += p1r[2][j]/3.f;
    A[1][5+j] += p1r[3][j]/3.f; A[5+j][1] += p1r[3][j]/3.f;
  }
  A[1][1] += p0[2];
  #pragma unroll
  for (int i = 0; i < 3; ++i) { A[2+i][2+i] += p0[3]/3.f; A[5+i][5+i] += p0[5]/3.f; }
  // Wigner 1x1->2 blocks: W[i][j] = C2 * sum_k M2[k,i,j] q[k]
  #pragma unroll
  for (int blk = 0; blk < 3; ++blk) {
    const float* q = p2r[blk];
    float W[3][3];
    W[0][0] = C2*(-S6*q[2] - S2*q[4]);
    W[1][1] = C2*( 2.f*S6*q[2]);
    W[2][2] = C2*(-S6*q[2] + S2*q[4]);
    W[0][1] = W[1][0] = C2*S2*q[1];
    W[0][2] = W[2][0] = C2*S2*q[0];
    W[1][2] = W[2][1] = C2*S2*q[3];
    #pragma unroll
    for (int i = 0; i < 3; ++i)
      #pragma unroll
      for (int j = 0; j < 3; ++j) {
        if (blk == 0) A[2+i][2+j] += W[i][j];
        if (blk == 2) A[5+i][5+j] += W[i][j];
        if (blk == 1) { A[2+i][5+j] += W[i][j]; A[5+j][2+i] += W[i][j]; }
      }
  }
  #pragma unroll
  for (int i = 0; i < 3; ++i)
    #pragma unroll
    for (int j = 0; j < 3; ++j)
      if (i == j) { A[2+i][5+j] += p0[4]/3.f; A[5+j][2+i] += p0[4]/3.f; }

  // cyclic Jacobi eigensolver, columns of V are eigenvectors
  float V[8][8];
  #pragma unroll
  for (int i = 0; i < 8; ++i)
    #pragma unroll
    for (int j = 0; j < 8; ++j) V[i][j] = (i == j) ? 1.f : 0.f;

  #pragma unroll 1
  for (int sweep = 0; sweep < 8; ++sweep) {
    #pragma unroll
    for (int p = 0; p < 7; ++p) {
      #pragma unroll
      for (int q = p + 1; q < 8; ++q) {
        float apq = A[p][q];
        float app = A[p][p], aqq = A[q][q];
        float denom = 2.f * apq;
        float th = (aqq - app) *
                   __builtin_amdgcn_rcpf(denom + copysignf(1e-30f, denom));
        float t  = copysignf(1.f, th) *
                   __builtin_amdgcn_rcpf(fabsf(th) + sqrtf(1.f + th * th));
        float c  = __builtin_amdgcn_rsqf(1.f + t * t);   // v_rsq_f32
        float s  = t * c;
        if (fabsf(apq) < 1e-12f) { c = 1.f; s = 0.f; }
        #pragma unroll
        for (int k = 0; k < 8; ++k) {
          float t1 = A[p][k], t2 = A[q][k];
          A[p][k] = c * t1 - s * t2;
          A[q][k] = s * t1 + c * t2;
        }
        #pragma unroll
        for (int k = 0; k < 8; ++k) {
          float t1 = A[k][p], t2 = A[k][q];
          A[k][p] = c * t1 - s * t2;
          A[k][q] = s * t1 + c * t2;
        }
        #pragma unroll
        for (int k = 0; k < 8; ++k) {
          float t1 = V[k][p], t2 = V[k][q];
          V[k][p] = c * t1 - s * t2;
          V[k][q] = s * t1 + c * t2;
        }
      }
    }
  }

  // rank eigenvalues ascending; output 4 smallest eigenvectors as rows
  float d[8];
  #pragma unroll
  for (int j = 0; j < 8; ++j) d[j] = A[j][j];
  int rank[8];
  #pragma unroll
  for (int j = 0; j < 8; ++j) {
    int r = 0;
    #pragma unroll
    for (int k = 0; k < 8; ++k)
      r += (d[k] < d[j]) || (d[k] == d[j] && k < j);
    rank[j] = r;
  }
  #pragma unroll
  for (int p = 0; p < 4; ++p)
    #pragma unroll
    for (int i = 0; i < 8; ++i) {
      float val = 0.f;
      #pragma unroll
      for (int j = 0; j < 8; ++j)
        val += (rank[j] == p) ? V[i][j] : 0.f;
      out[b * 32 + p * 8 + i] = val;
    }
}

extern "C" void kernel_launch(void* const* d_in, const int* in_sizes, int n_in,
                              void* d_out, int out_size, void* d_ws, size_t ws_size,
                              hipStream_t stream) {
  const float* pos = (const float*)d_in[0];
  const float* xc  = (const float*)d_in[1];
  const float* zin = (const float*)d_in[2];
  const float* fw1 = (const float*)d_in[3];
  const float* fw2 = (const float*)d_in[4];
  const float* w00 = (const float*)d_in[5];
  const float* w11 = (const float*)d_in[6];
  const float* w01 = (const float*)d_in[7];
  const float* w10 = (const float*)d_in[8];
  const float* w2  = (const float*)d_in[9];
  float* aBuf = (float*)d_ws;          // BATCH*20 floats = 2.6 MB
  float* out  = (float*)d_out;

  k_edge<<<BATCH / 8, 256, 0, stream>>>(pos, xc, zin, fw1, fw2, aBuf);
  k_batch<<<BATCH / 64, 64, 0, stream>>>(aBuf, w00, w11, w01, w10, w2, out);
}
